// HierarchicalMILRegressor_48971217109620
// MI455X (gfx1250) — matrix operand order repaired and checked by reading, not compiled, not run on hardware
//
#include <hip/hip_runtime.h>
#include <hip/hip_bf16.h>

typedef __attribute__((ext_vector_type(16))) __bf16 v16bf;
typedef __attribute__((ext_vector_type(8)))  __bf16 v8bf;
typedef __attribute__((ext_vector_type(8)))  float  v8f;

#define D_EMB   256
#define KDIM    2048
#define MROWS   16384
#define HID     128
#define NTRIAL  512     // B*A*R
#define NBA     32      // B*A
#define NBATCH  16
#define BS_STRIDE 56    // padded LDS row stride (bf16 elems): 112 B, bank-conflict-free

// ---------------------------------------------------------------------------
// Kernel 0: W_enc (2048x256 f32, row-major) -> Wt (256x2048 bf16, N-major)
// ---------------------------------------------------------------------------
__global__ __launch_bounds__(256) void wenc_convert_kernel(
    const float* __restrict__ W, __bf16* __restrict__ Wt)
{
    const int tid = blockIdx.x * 256 + threadIdx.x;   // tid = n*2048 + k
    const int n = tid >> 11;
    const int k = tid & 2047;
    Wt[tid] = (__bf16)W[(size_t)k * D_EMB + n];       // coalesced write
}

// ---------------------------------------------------------------------------
// Kernel 1: h = windows(16384x2048) @ W_enc + b_enc  via v_wmma_f32_16x16x32_bf16
// block = 256 threads = 8 waves, 128 rows x 256 cols per block.
// B K-slice staged in LDS (double-buffered); A raw chunk double-buffered in regs.
// ---------------------------------------------------------------------------
__global__ __launch_bounds__(256) void encoder_gemm_kernel(
    const float*  __restrict__ A,     // 16384 x 2048 f32
    const __bf16* __restrict__ Bt,    // 256 x 2048 bf16 (N-major)
    const float*  __restrict__ bias,  // 256
    float*        __restrict__ H)     // 16384 x 256 f32
{
    __shared__ __bf16 Bs[2][256 * BS_STRIDE];         // 2 x 28 KB

    const int tid     = threadIdx.x;
    const int wave    = tid >> 5;
    const int lane    = tid & 31;
    const int lmod    = lane & 15;
    const int lhigh   = lane >> 4;                    // 0 or 1
    const int rowbase = blockIdx.x * 128 + wave * 16;

    // A fragment per-lane base: row = rowbase+lmod; K chunks at 8*lhigh and 16+8*lhigh
    const float*  arow = A  + (size_t)(rowbase + lmod) * KDIM + 8 * lhigh;
    // B staging: thread tid owns N-row n = tid
    const __bf16* brow = Bt + (size_t)tid * KDIM;

    v8f acc[16];
#pragma unroll
    for (int i = 0; i < 16; ++i) acc[i] = (v8f)0.0f;

    // ---- prologue: stage B slice 0, load A chunk 0 ----
    {
        const __bf16* gp = brow;                      // ks = 0
        uint4 b0 = *(const uint4*)(gp);
        uint4 b1 = *(const uint4*)(gp + 8);
        uint4 b2 = *(const uint4*)(gp + 16);
        uint4 b3 = *(const uint4*)(gp + 24);
        __bf16* lp = &Bs[0][tid * BS_STRIDE];
        *(uint4*)(lp)      = b0;
        *(uint4*)(lp + 8)  = b1;
        *(uint4*)(lp + 16) = b2;
        *(uint4*)(lp + 24) = b3;
    }
    float4 c0 = *(const float4*)(arow);
    float4 c1 = *(const float4*)(arow + 4);
    float4 c2 = *(const float4*)(arow + 16);
    float4 c3 = *(const float4*)(arow + 20);
    __syncthreads();

    int buf = 0;
    for (int ks = 0; ks < KDIM; ks += 32) {
        const bool more = (ks + 32) < KDIM;
        float4 n0, n1, n2, n3;
        if (more) {
            // prefetch A for step i+2 (primes L2), load raw A for step i+1
            const float* ap = arow + ks + 32;
            __builtin_prefetch(ap + 32, 0, 1);
            n0 = *(const float4*)(ap);
            n1 = *(const float4*)(ap + 4);
            n2 = *(const float4*)(ap + 16);
            n3 = *(const float4*)(ap + 20);
            // stage next B slice into the other LDS buffer
            const __bf16* gp = brow + ks + 32;
            uint4 b0 = *(const uint4*)(gp);
            uint4 b1 = *(const uint4*)(gp + 8);
            uint4 b2 = *(const uint4*)(gp + 16);
            uint4 b3 = *(const uint4*)(gp + 24);
            __bf16* lp = &Bs[buf ^ 1][tid * BS_STRIDE];
            *(uint4*)(lp)      = b0;
            *(uint4*)(lp + 8)  = b1;
            *(uint4*)(lp + 16) = b2;
            *(uint4*)(lp + 24) = b3;
        }

        // convert current A chunk to bf16 fragment
        v16bf af;
        af[0]  = (__bf16)c0.x; af[1]  = (__bf16)c0.y; af[2]  = (__bf16)c0.z; af[3]  = (__bf16)c0.w;
        af[4]  = (__bf16)c1.x; af[5]  = (__bf16)c1.y; af[6]  = (__bf16)c1.z; af[7]  = (__bf16)c1.w;
        af[8]  = (__bf16)c2.x; af[9]  = (__bf16)c2.y; af[10] = (__bf16)c2.z; af[11] = (__bf16)c2.w;
        af[12] = (__bf16)c3.x; af[13] = (__bf16)c3.y; af[14] = (__bf16)c3.z; af[15] = (__bf16)c3.w;

        const __bf16* bbase = &Bs[buf][lmod * BS_STRIDE + 8 * lhigh];
#pragma unroll
        for (int nt = 0; nt < 16; ++nt) {
            const __bf16* fp = bbase + nt * 16 * BS_STRIDE;
            v8bf blo = *(const v8bf*)(fp);
            v8bf bhi = *(const v8bf*)(fp + 16);
            v16bf bfrag = __builtin_shufflevector(blo, bhi,
                0, 1, 2, 3, 4, 5, 6, 7, 8, 9, 10, 11, 12, 13, 14, 15);
            acc[nt] = __builtin_amdgcn_wmma_f32_16x16x32_bf16(
                false, af, false, bfrag, (short)0, acc[nt], false, false);
        }

        __syncthreads();   // waves done with Bs[buf]; Bs[buf^1] stores complete
        if (more) {
            c0 = n0; c1 = n1; c2 = n2; c3 = n3;
            buf ^= 1;
        }
    }

    // Epilogue: C/D layout — lane half selects M offset 8, VGPR v = row within half
#pragma unroll
    for (int nt = 0; nt < 16; ++nt) {
        const int   ncol = nt * 16 + lmod;
        const float bv   = bias[ncol];
#pragma unroll
        for (int v = 0; v < 8; ++v) {
            const int m = rowbase + lhigh * 8 + v;
            H[(size_t)m * D_EMB + ncol] = acc[nt][v] + bv;
        }
    }
}

// ---------------------------------------------------------------------------
// Kernel 2: gated attention score per row: out = (tanh(xV+vb)*sigmoid(xU+ub)) @ w + wb
// one block (128 threads) per row of X (row width 256)
// ---------------------------------------------------------------------------
__global__ __launch_bounds__(128) void gated_score_kernel(
    const float* __restrict__ X,
    const float* __restrict__ Vw, const float* __restrict__ Vb,
    const float* __restrict__ Uw, const float* __restrict__ Ub,
    const float* __restrict__ ww, const float* __restrict__ wb,
    float* __restrict__ out)
{
    __shared__ float xrow[D_EMB];
    __shared__ float red[HID];
    const int row = blockIdx.x;
    const int j   = threadIdx.x;

    xrow[j]       = X[(size_t)row * D_EMB + j];
    xrow[j + 128] = X[(size_t)row * D_EMB + j + 128];
    __syncthreads();

    float av = Vb[j], au = Ub[j];
    for (int k = 0; k < D_EMB; ++k) {
        const float x = xrow[k];
        av = fmaf(x, Vw[(size_t)k * HID + j], av);    // coalesced over j
        au = fmaf(x, Uw[(size_t)k * HID + j], au);
    }
    const float g = tanhf(av) * (1.0f / (1.0f + expf(-au)));
    red[j] = g * ww[j];
    __syncthreads();
    for (int s = 64; s > 0; s >>= 1) {
        if (j < s) red[j] += red[j + s];
        __syncthreads();
    }
    if (j == 0) out[row] = red[0] + wb[0];
}

// ---------------------------------------------------------------------------
// Kernel 3: masked softmax over K=32 windows + pool -> trial_emb (512 x 256)
// ---------------------------------------------------------------------------
__global__ __launch_bounds__(256) void window_pool_kernel(
    const float* __restrict__ h, const float* __restrict__ logits,
    const unsigned char* __restrict__ mask, float* __restrict__ trial_emb)
{
    __shared__ float salpha[32];
    const int trial = blockIdx.x;
    const int t     = threadIdx.x;
    if (t == 0) {
        float l[32]; int mk[32];
        float mx = -3.4e38f;
        for (int k = 0; k < 32; ++k) {
            mk[k] = mask[trial * 32 + k] != 0;
            l[k]  = mk[k] ? logits[trial * 32 + k] : -1e9f;
            mx    = fmaxf(mx, l[k]);
        }
        float s = 0.0f;
        for (int k = 0; k < 32; ++k) { l[k] = expf(l[k] - mx); s += l[k]; }
        const float inv = 1.0f / s;
        for (int k = 0; k < 32; ++k) salpha[k] = mk[k] ? l[k] * inv : 0.0f;
    }
    __syncthreads();
    float acc = 0.0f;
    for (int k = 0; k < 32; ++k)
        acc = fmaf(salpha[k], h[(size_t)(trial * 32 + k) * D_EMB + t], acc);
    trial_emb[(size_t)trial * D_EMB + t] = acc;
}

// ---------------------------------------------------------------------------
// Kernel 4: masked softmax over R=16 trials + pool -> axis_emb (32 x 256)
// ---------------------------------------------------------------------------
__global__ __launch_bounds__(256) void trial_pool_kernel(
    const float* __restrict__ trial_emb, const float* __restrict__ logits,
    const unsigned char* __restrict__ mask, float* __restrict__ axis_emb)
{
    __shared__ float sbeta[16];
    const int ba = blockIdx.x;     // 0..31
    const int t  = threadIdx.x;
    if (t == 0) {
        float l[16]; int mk[16];
        float mx = -3.4e38f;
        for (int r = 0; r < 16; ++r) {
            mk[r] = mask[ba * 16 + r] != 0;
            l[r]  = mk[r] ? logits[ba * 16 + r] : -1e9f;
            mx    = fmaxf(mx, l[r]);
        }
        float s = 0.0f;
        for (int r = 0; r < 16; ++r) { l[r] = expf(l[r] - mx); s += l[r]; }
        const float inv = 1.0f / s;
        for (int r = 0; r < 16; ++r) sbeta[r] = mk[r] ? l[r] * inv : 0.0f;
    }
    __syncthreads();
    float acc = 0.0f;
    for (int r = 0; r < 16; ++r)
        acc = fmaf(sbeta[r], trial_emb[(size_t)(ba * 16 + r) * D_EMB + t], acc);
    axis_emb[(size_t)ba * D_EMB + t] = acc;
}

// ---------------------------------------------------------------------------
// Kernel 5: fusion MLP head, one block (256 threads) per batch element
// ---------------------------------------------------------------------------
__device__ __forceinline__ float gelu_exact(float x) {
    return 0.5f * x * (1.0f + erff(x * 0.70710678118654752f));
}

__global__ __launch_bounds__(256) void head_kernel(
    const float* __restrict__ axis_emb,
    const float* __restrict__ f1w, const float* __restrict__ f1b,
    const float* __restrict__ f2w, const float* __restrict__ f2b,
    const float* __restrict__ hwhw, const float* __restrict__ hwhb,
    const float* __restrict__ hsew, const float* __restrict__ hseb,
    float* __restrict__ out)
{
    __shared__ float z[512];
    __shared__ float z1[256];
    __shared__ float z2[256];
    __shared__ float red[256];
    const int b = blockIdx.x;
    const int j = threadIdx.x;

    z[j]       = axis_emb[(size_t)(b * 2 + 0) * D_EMB + j];
    z[256 + j] = axis_emb[(size_t)(b * 2 + 1) * D_EMB + j];
    __syncthreads();

    float a = f1b[j];
    for (int i = 0; i < 512; ++i) a = fmaf(z[i], f1w[(size_t)i * 256 + j], a);
    z1[j] = gelu_exact(a);
    __syncthreads();

    float c = f2b[j];
    for (int i = 0; i < 256; ++i) c = fmaf(z1[i], f2w[(size_t)i * 256 + j], c);
    z2[j] = gelu_exact(c);
    __syncthreads();

    red[j] = z2[j] * hwhw[j];
    __syncthreads();
    for (int s = 128; s > 0; s >>= 1) { if (j < s) red[j] += red[j + s]; __syncthreads(); }
    if (j == 0) out[b] = 24.0f / (1.0f + expf(-(red[0] + hwhb[0])));
    __syncthreads();

    red[j] = z2[j] * hsew[j];
    __syncthreads();
    for (int s = 128; s > 0; s >>= 1) { if (j < s) red[j] += red[j + s]; __syncthreads(); }
    if (j == 0) out[16 + b] = 42.0f / (1.0f + expf(-(red[0] + hseb[0])));
}

// ---------------------------------------------------------------------------
extern "C" void kernel_launch(void* const* d_in, const int* in_sizes, int n_in,
                              void* d_out, int out_size, void* d_ws, size_t ws_size,
                              hipStream_t stream) {
    (void)in_sizes; (void)n_in; (void)out_size; (void)ws_size;

    const float*         windows = (const float*)d_in[0];
    const unsigned char* wmask   = (const unsigned char*)d_in[1]; // bool, 1B/elem
    const unsigned char* tmask   = (const unsigned char*)d_in[2];
    const float* W_enc = (const float*)d_in[3];
    const float* b_enc = (const float*)d_in[4];
    const float* Vw_w  = (const float*)d_in[5];
    const float* Vw_b  = (const float*)d_in[6];
    const float* Uw_w  = (const float*)d_in[7];
    const float* Uw_b  = (const float*)d_in[8];
    const float* ww_w  = (const float*)d_in[9];
    const float* ww_b  = (const float*)d_in[10];
    const float* Vt_w  = (const float*)d_in[11];
    const float* Vt_b  = (const float*)d_in[12];
    const float* Ut_w  = (const float*)d_in[13];
    const float* Ut_b  = (const float*)d_in[14];
    const float* wt_w  = (const float*)d_in[15];
    const float* wt_b  = (const float*)d_in[16];
    const float* f1_w  = (const float*)d_in[17];
    const float* f1_b  = (const float*)d_in[18];
    const float* f2_w  = (const float*)d_in[19];
    const float* f2_b  = (const float*)d_in[20];
    const float* hwh_w = (const float*)d_in[21];
    const float* hwh_b = (const float*)d_in[22];
    const float* hse_w = (const float*)d_in[23];
    const float* hse_b = (const float*)d_in[24];

    float* out = (float*)d_out;

    // workspace layout (bytes)
    char*   ws   = (char*)d_ws;
    __bf16* Wt   = (__bf16*)ws;                             //   1,048,576
    float*  h    = (float*)(ws + (1u << 20));               //  16,777,216
    float*  wl   = (float*)(ws + (1u << 20) + (16u << 20)); //      65,536
    float*  temb = wl + MROWS;                              //     524,288
    float*  tl   = temb + NTRIAL * D_EMB;                   //       2,048
    float*  aemb = tl + NTRIAL;                             //      32,768

    // 0) W_enc -> bf16 transposed
    wenc_convert_kernel<<<dim3((256 * KDIM) / 256), dim3(256), 0, stream>>>(W_enc, Wt);
    // 1) encoder GEMM (WMMA bf16, LDS double-buffered B): h = windows @ W_enc + b_enc
    encoder_gemm_kernel<<<dim3(MROWS / 128), dim3(256), 0, stream>>>(windows, Wt, b_enc, h);
    // 2) window-level gated scores (16384 rows)
    gated_score_kernel<<<dim3(MROWS), dim3(HID), 0, stream>>>(
        h, Vw_w, Vw_b, Uw_w, Uw_b, ww_w, ww_b, wl);
    // 3) masked softmax over K + pool -> trial_emb
    window_pool_kernel<<<dim3(NTRIAL), dim3(256), 0, stream>>>(h, wl, wmask, temb);
    // 4) trial-level gated scores (512 rows)
    gated_score_kernel<<<dim3(NTRIAL), dim3(HID), 0, stream>>>(
        temb, Vt_w, Vt_b, Ut_w, Ut_b, wt_w, wt_b, tl);
    // 5) masked softmax over R + pool -> axis_emb
    trial_pool_kernel<<<dim3(NBA), dim3(256), 0, stream>>>(temb, tl, tmask, aemb);
    // 6) fusion head -> out[0:16]=wh, out[16:32]=se
    head_kernel<<<dim3(NBATCH), dim3(256), 0, stream>>>(
        aemb, f1_w, f1_b, f2_w, f2_b, hwh_w, hwh_b, hse_w, hse_b, out);
}
